// ImprovedMesaNetLayer_15075335209582
// MI455X (gfx1250) — compile-verified, matrix operand order checked by reading, not run on hardware
//
#include <hip/hip_runtime.h>
#include <cstddef>

#define DD 512
#define HH 256
#define QQ 128
#define AA 32

typedef __attribute__((ext_vector_type(16))) __bf16 v16bf;
typedef __attribute__((ext_vector_type(8)))  float  v8f;
typedef __attribute__((ext_vector_type(8)))  unsigned short ush8;

__device__ __forceinline__ unsigned short f2bf(float f) {
  unsigned u = __builtin_bit_cast(unsigned, f);
  return (unsigned short)((u + 0x7FFFu + ((u >> 16) & 1u)) >> 16);  // RNE f32->bf16
}
__device__ __forceinline__ float bf2f(unsigned short h) {
  unsigned u = ((unsigned)h) << 16;
  return __builtin_bit_cast(float, u);
}

// CDNA5 async copy: 16B global -> LDS, tracked by ASYNCcnt (ISA 15.18.3 op 98).
#define ASYNC_CP16(lds_off, gptr)                                         \
  asm volatile("global_load_async_to_lds_b128 %0, %1, off" ::"v"(lds_off), \
               "v"(gptr)                                                   \
               : "memory")
#define ASYNC_WAIT() asm volatile("s_wait_asynccnt 0x0" ::: "memory")

// WMMA bf16 fragment from an LDS tile row (row-major, K-window of 32).
// ISA 7.12.2: idx=lane%16 is row(A)/col(B); half=lane/16 picks k in
// {8h..8h+7} (VGPR0-3) and {16+8h..+7} (VGPR4-7): two contiguous 16B ds loads.
__device__ __forceinline__ v16bf load_frag(const unsigned short* row, int koff) {
  union { v16bf v; ush8 h[2]; } u;
  u.h[0] = *(const ush8*)(row + koff);
  u.h[1] = *(const ush8*)(row + 16 + koff);
  return u.v;
}

// ---------------------------------------------------------------------------
// One-shot f32 -> bf16 cast (handles strided source, e.g. r_w1 with ld=513).
// ---------------------------------------------------------------------------
__global__ __launch_bounds__(256) void cast2d_bf16(const float* __restrict__ src,
                                                   unsigned short* __restrict__ dst,
                                                   int rows, int cols, int ld) {
  size_t i = (size_t)blockIdx.x * 256 + threadIdx.x;
  size_t total = (size_t)rows * cols / 4;
  if (i >= total) return;
  size_t e = i * 4;
  int r = (int)(e / (size_t)cols), c = (int)(e % (size_t)cols);
  const float* sp = src + (size_t)r * ld + c;
  float a0, a1, a2, a3;
  if ((ld & 3) == 0) {
    float4 v = *(const float4*)sp;
    a0 = v.x; a1 = v.y; a2 = v.z; a3 = v.w;
  } else {
    a0 = sp[0]; a1 = sp[1]; a2 = sp[2]; a3 = sp[3];
  }
  ushort4 o;
  o.x = f2bf(a0); o.y = f2bf(a1); o.z = f2bf(a2); o.w = f2bf(a3);
  *(ushort4*)(dst + e) = o;
}

// ---------------------------------------------------------------------------
// Fused GEMM: Y[M,BN] = epi( Xb[M,KD](bf16) @ Wb[BN,KD](bf16)^T + bias )
// EPI: 0 bias, 1 bias+LN+ReLU, 2 bias+ReLU, 3 bias+rank1(energy)+LN+ReLU
// OUTF32: write f32 (final) else bf16 (feeds next WMMA stage).
// Block: 32 rows x BN cols, 256 threads = 8 waves (2 M x 4 N). Tiles staged
// with GLOBAL_LOAD_ASYNC_TO_LDS_B128 + s_wait_asynccnt.
// ---------------------------------------------------------------------------
template <int BN, int KD, int EPI, bool OUTF32>
__global__ __launch_bounds__(256) void gemm_fused(
    const unsigned short* __restrict__ Xb, const unsigned short* __restrict__ Wb,
    const float* __restrict__ bias, const float* __restrict__ gamma,
    const float* __restrict__ beta, const float* __restrict__ evec,
    const float* __restrict__ wfull, int ldwf, void* __restrict__ Yraw) {
  constexpr int BM = 32, KT = 32, NT = BN / 64;
  __shared__ alignas(16) unsigned short la[BM * KT];
  __shared__ alignas(16) unsigned short lb[BN * KT];
  __shared__ float lsum[BM * 4];
  __shared__ float lsq[BM * 4];
  __shared__ float le[BM];
  const int tid = threadIdx.x, wave = tid >> 5, lane = tid & 31;
  const int idx = lane & 15, half = lane >> 4;
  const int wm = wave & 1, wn = wave >> 1;
  const int m0 = blockIdx.x * BM;
  const unsigned la_base = (unsigned)(size_t)&la[0];
  const unsigned lb_base = (unsigned)(size_t)&lb[0];

  if constexpr (EPI == 3) {
    if (tid < BM) le[tid] = evec[m0 + tid];
  }
  v8f acc[NT] = {};

  for (int k0 = 0; k0 < KD; k0 += KT) {
    __syncthreads();
    for (int i = tid; i < BM * 4; i += 256) {
      int r = i >> 2, c = (i & 3) << 3;
      unsigned lo = la_base + (unsigned)(((r << 5) + c) << 1);
      const unsigned short* gp = Xb + (size_t)(m0 + r) * KD + k0 + c;
      ASYNC_CP16(lo, gp);
    }
    for (int i = tid; i < BN * 4; i += 256) {
      int r = i >> 2, c = (i & 3) << 3;
      unsigned lo = lb_base + (unsigned)(((r << 5) + c) << 1);
      const unsigned short* gp = Wb + (size_t)r * KD + k0 + c;
      ASYNC_CP16(lo, gp);
    }
    ASYNC_WAIT();
    __syncthreads();
    v16bf af = load_frag(&la[(wm * 16 + idx) * KT], 8 * half);
#pragma unroll
    for (int t = 0; t < NT; ++t) {
      int col = wn * (NT * 16) + t * 16 + idx;
      v16bf bv = load_frag(&lb[col * KT], 8 * half);
      acc[t] = __builtin_amdgcn_wmma_f32_16x16x32_bf16(false, af, false, bv,
                                                       (short)0, acc[t], false, false);
    }
  }

  // bias (+rank1) (+relu for EPI==2)
#pragma unroll
  for (int t = 0; t < NT; ++t) {
    int col = wn * (NT * 16) + t * 16 + idx;
    float bv = bias[col];
    float wl = 0.0f;
    if constexpr (EPI == 3) wl = wfull[(size_t)col * ldwf + KD];  // r_w1[:,512]
    (void)wl;
#pragma unroll
    for (int r = 0; r < 8; ++r) {
      float v = acc[t][r] + bv;
      if constexpr (EPI == 3) v += le[wm * 16 + half * 8 + r] * wl;
      if constexpr (EPI == 2) v = fmaxf(v, 0.0f);
      acc[t][r] = v;
    }
  }

  if constexpr (EPI == 1 || EPI == 3) {  // LayerNorm over full row + ReLU
    float s[8], q[8];
#pragma unroll
    for (int r = 0; r < 8; ++r) { s[r] = 0.f; q[r] = 0.f; }
#pragma unroll
    for (int t = 0; t < NT; ++t)
#pragma unroll
      for (int r = 0; r < 8; ++r) {
        float v = acc[t][r];
        s[r] += v; q[r] += v * v;
      }
#pragma unroll
    for (int r = 0; r < 8; ++r)
      for (int off = 1; off < 16; off <<= 1) {
        s[r] += __shfl_xor(s[r], off, 32);
        q[r] += __shfl_xor(q[r], off, 32);
      }
    if (idx == 0) {
#pragma unroll
      for (int r = 0; r < 8; ++r) {
        int m = wm * 16 + half * 8 + r;
        lsum[m * 4 + wn] = s[r];
        lsq[m * 4 + wn] = q[r];
      }
    }
    __syncthreads();
#pragma unroll
    for (int r = 0; r < 8; ++r) {
      int m = wm * 16 + half * 8 + r;
      float sum = lsum[m * 4] + lsum[m * 4 + 1] + lsum[m * 4 + 2] + lsum[m * 4 + 3];
      float sq  = lsq[m * 4]  + lsq[m * 4 + 1]  + lsq[m * 4 + 2]  + lsq[m * 4 + 3];
      float mean = sum * (1.0f / BN);
      float var  = sq * (1.0f / BN) - mean * mean;
      float rstd = rsqrtf(var + 1e-5f);
#pragma unroll
      for (int t = 0; t < NT; ++t) {
        int col = wn * (NT * 16) + t * 16 + idx;
        float v = (acc[t][r] - mean) * rstd * gamma[col] + beta[col];
        acc[t][r] = fmaxf(v, 0.0f);
      }
    }
  }

#pragma unroll
  for (int t = 0; t < NT; ++t) {
    int col = wn * (NT * 16) + t * 16 + idx;
#pragma unroll
    for (int r = 0; r < 8; ++r) {
      int m = wm * 16 + half * 8 + r;
      if constexpr (OUTF32)
        ((float*)Yraw)[(size_t)(m0 + m) * BN + col] = acc[t][r];
      else
        ((unsigned short*)Yraw)[(size_t)(m0 + m) * BN + col] = f2bf(acc[t][r]);
    }
  }
}

// energy = tanh(H3b(bf16) @ e_w4^T + e_b4)
__global__ void energy_head(const unsigned short* __restrict__ H3b,
                            const float* __restrict__ w4,
                            const float* __restrict__ b4, float* __restrict__ E, int M) {
  int n = blockIdx.x * blockDim.x + threadIdx.x;
  if (n >= M) return;
  const unsigned short* h = H3b + (size_t)n * QQ;
  float s = 0.f;
  for (int k = 0; k < QQ; ++k) s += bf2f(h[k]) * w4[k];
  E[n] = tanhf(s + b4[0]);
}

// temp = 1 + mean(|energy|)
__global__ __launch_bounds__(1024) void temp_reduce(const float* __restrict__ E,
                                                    float* __restrict__ T, int M) {
  __shared__ float red[1024];
  int tid = threadIdx.x;
  float s = 0.f;
  for (int i = tid; i < M; i += 1024) s += fabsf(E[i]);
  red[tid] = s;
  __syncthreads();
  for (int off = 512; off > 0; off >>= 1) {
    if (tid < off) red[tid] += red[tid + off];
    __syncthreads();
  }
  if (tid == 0) T[0] = 1.0f + red[0] / (float)M;
}

// logits = R2b @ r_w3^T + r_b3 ; probs = softmax(logits / temp). 32 tokens/block.
__global__ __launch_bounds__(128) void probs_softmax(
    const unsigned short* __restrict__ R2b, const unsigned short* __restrict__ W3b,
    const float* __restrict__ bias, const float* __restrict__ tempv,
    float* __restrict__ P) {
  __shared__ alignas(16) unsigned short la[32 * 32];
  __shared__ alignas(16) unsigned short lb[32 * 32];
  __shared__ float lt[32 * 32];
  const int tid = threadIdx.x, wave = tid >> 5, lane = tid & 31;
  const int idx = lane & 15, half = lane >> 4;
  const int wm = wave & 1, wn = wave >> 1;
  const int m0 = blockIdx.x * 32;
  const unsigned la_base = (unsigned)(size_t)&la[0];
  const unsigned lb_base = (unsigned)(size_t)&lb[0];
  v8f acc = {};
  for (int k0 = 0; k0 < HH; k0 += 32) {
    __syncthreads();
    for (int i = tid; i < 32 * 4; i += 128) {
      int r = i >> 2, c = (i & 3) << 3;
      unsigned off = (unsigned)(((r << 5) + c) << 1);
      ASYNC_CP16(la_base + off, R2b + (size_t)(m0 + r) * HH + k0 + c);
      ASYNC_CP16(lb_base + off, W3b + (size_t)r * HH + k0 + c);
    }
    ASYNC_WAIT();
    __syncthreads();
    v16bf af = load_frag(&la[(wm * 16 + idx) * 32], 8 * half);
    v16bf bv = load_frag(&lb[(wn * 16 + idx) * 32], 8 * half);
    acc = __builtin_amdgcn_wmma_f32_16x16x32_bf16(false, af, false, bv, (short)0,
                                                  acc, false, false);
  }
  float invt = 1.0f / tempv[0];
#pragma unroll
  for (int r = 0; r < 8; ++r) {
    int m = wm * 16 + half * 8 + r, col = wn * 16 + idx;
    lt[m * 32 + col] = (acc[r] + bias[col]) * invt;
  }
  __syncthreads();
  if (tid < 32) {
    float mx = -1e30f;
    for (int j = 0; j < AA; ++j) mx = fmaxf(mx, lt[tid * 32 + j]);
    float s = 0.f, e[AA];
    for (int j = 0; j < AA; ++j) { e[j] = __expf(lt[tid * 32 + j] - mx); s += e[j]; }
    float inv = 1.0f / s;
    for (int j = 0; j < AA; ++j) P[(size_t)(m0 + tid) * AA + j] = e[j] * inv;
  }
}

// const[a,d] = aW[a,d,:] . anchor_emb[a,:] + ab[a,d]   (f32, tiny)
__global__ __launch_bounds__(256) void anchor_const(
    const float* __restrict__ aW, const float* __restrict__ aemb,
    const float* __restrict__ ab, float* __restrict__ C) {
  int i = blockIdx.x * 256 + threadIdx.x;
  if (i >= AA * DD) return;
  int a = i / DD, d = i % DD;
  const float* w = aW + ((size_t)a * DD + d) * DD;
  const float* e = aemb + (size_t)a * DD;
  float s = 0.f;
  for (int k = 0; k < DD; ++k) s += w[k] * e[k];
  C[i] = s + ab[i];
}

// ---------------------------------------------------------------------------
// RXb[n,d] = bf16( x[n,d] + sum_a probs[n,a]*((x @ aW[a]^T)[d] + const[a,d]) )
// Dominant kernel (34.4 G MACs). x tile persists in LDS across all 32 anchors
// per K-half; all staging via async-to-LDS B128.
// ---------------------------------------------------------------------------
__global__ __launch_bounds__(256) void routed_fused(
    const unsigned short* __restrict__ Xb, const float* __restrict__ Xf,
    const unsigned short* __restrict__ aWb, const float* __restrict__ P,
    const float* __restrict__ C, unsigned short* __restrict__ Y) {
  constexpr int BM = 32, KT = 32, NT = 8, KH = 256;
  __shared__ alignas(16) unsigned short xt[BM * KH];  // 16 KB
  __shared__ alignas(16) unsigned short lb[DD * KT];  // 32 KB
  __shared__ float lp[BM * AA];                       //  4 KB
  const int tid = threadIdx.x, wave = tid >> 5, lane = tid & 31;
  const int idx = lane & 15, half = lane >> 4;
  const int wm = wave & 1, wn = wave >> 1;
  const int m0 = blockIdx.x * BM;
  const unsigned xt_base = (unsigned)(size_t)&xt[0];
  const unsigned lb_base = (unsigned)(size_t)&lb[0];
  v8f routed[NT] = {};

  for (int kh = 0; kh < DD; kh += KH) {
    __syncthreads();
    for (int i = tid; i < BM * (KH / 8); i += 256) {  // 1024 chunks
      int r = i >> 5, c = (i & 31) << 3;
      unsigned lo = xt_base + (unsigned)((r * KH + c) << 1);
      ASYNC_CP16(lo, Xb + (size_t)(m0 + r) * DD + kh + c);
    }
    if (kh == 0)
      for (int i = tid; i < BM * AA; i += 256)
        lp[i] = P[(size_t)(m0 + (i >> 5)) * AA + (i & 31)];
    ASYNC_WAIT();
    __syncthreads();

    for (int a = 0; a < AA; ++a) {
      const unsigned short* Wp = aWb + (size_t)a * DD * DD;
      v8f part[NT] = {};
      for (int k0 = 0; k0 < KH; k0 += KT) {
        __syncthreads();
        for (int i = tid; i < DD * 4; i += 256) {
          int r = i >> 2, c = (i & 3) << 3;
          unsigned lo = lb_base + (unsigned)(((r << 5) + c) << 1);
          ASYNC_CP16(lo, Wp + (size_t)r * DD + kh + k0 + c);
        }
        ASYNC_WAIT();
        __syncthreads();
        v16bf af = load_frag(&xt[(wm * 16 + idx) * KH + k0], 8 * half);
#pragma unroll
        for (int t = 0; t < NT; ++t) {
          int col = wn * 128 + t * 16 + idx;
          v16bf bv = load_frag(&lb[col * KT], 8 * half);
          part[t] = __builtin_amdgcn_wmma_f32_16x16x32_bf16(false, af, false, bv,
                                                            (short)0, part[t], false, false);
        }
      }
      float p[8];
#pragma unroll
      for (int r = 0; r < 8; ++r) p[r] = lp[(wm * 16 + half * 8 + r) * AA + a];
#pragma unroll
      for (int t = 0; t < NT; ++t) {
        int col = wn * 128 + t * 16 + idx;
        float cv = (kh == 0) ? C[a * DD + col] : 0.0f;
#pragma unroll
        for (int r = 0; r < 8; ++r) routed[t][r] += p[r] * (part[t][r] + cv);
      }
    }
  }

#pragma unroll
  for (int t = 0; t < NT; ++t) {
    int col = wn * 128 + t * 16 + idx;
#pragma unroll
    for (int r = 0; r < 8; ++r) {
      int m = wm * 16 + half * 8 + r;
      float v = routed[t][r] + Xf[(size_t)(m0 + m) * DD + col];
      Y[(size_t)(m0 + m) * DD + col] = f2bf(v);
    }
  }
}

// ---------------------------------------------------------------------------
extern "C" void kernel_launch(void* const* d_in, const int* in_sizes, int n_in,
                              void* d_out, int out_size, void* d_ws, size_t ws_size,
                              hipStream_t stream) {
  (void)n_in; (void)out_size; (void)ws_size;
  const float* x     = (const float*)d_in[0];
  const float* e_w1  = (const float*)d_in[1];
  const float* e_b1  = (const float*)d_in[2];
  const float* e_g1  = (const float*)d_in[3];
  const float* e_be1 = (const float*)d_in[4];
  const float* e_w2  = (const float*)d_in[5];
  const float* e_b2  = (const float*)d_in[6];
  const float* e_g2  = (const float*)d_in[7];
  const float* e_be2 = (const float*)d_in[8];
  const float* e_w3  = (const float*)d_in[9];
  const float* e_b3  = (const float*)d_in[10];
  const float* e_w4  = (const float*)d_in[11];
  const float* e_b4  = (const float*)d_in[12];
  const float* r_w1  = (const float*)d_in[13];
  const float* r_b1  = (const float*)d_in[14];
  const float* r_g1  = (const float*)d_in[15];
  const float* r_be1 = (const float*)d_in[16];
  const float* r_w2  = (const float*)d_in[17];
  const float* r_b2  = (const float*)d_in[18];
  const float* r_g2  = (const float*)d_in[19];
  const float* r_be2 = (const float*)d_in[20];
  const float* r_w3  = (const float*)d_in[21];
  const float* r_b3  = (const float*)d_in[22];
  const float* aemb  = (const float*)d_in[23];
  const float* aW    = (const float*)d_in[24];
  const float* ab    = (const float*)d_in[25];
  const float* o_w   = (const float*)d_in[26];
  const float* o_b   = (const float*)d_in[27];

  const int M  = in_sizes[0] / DD;  // 4096 tokens
  const int nb = M / 32;

  float* out    = (float*)d_out;
  float* probs  = out + (size_t)M * DD;
  float* energy = probs + (size_t)M * AA;

  // workspace carve-out (bf16 operand pool + f32 scratch)
  char* p = (char*)d_ws;
  auto aus = [&](size_t n) { unsigned short* r = (unsigned short*)p; p += n * 2; return r; };
  auto af32 = [&](size_t n) { float* r = (float*)p; p += n * 4; return r; };
  unsigned short* xb   = aus((size_t)M * DD);
  unsigned short* aWb  = aus((size_t)AA * DD * DD);
  unsigned short* ew1b = aus((size_t)DD * DD);
  unsigned short* ew2b = aus((size_t)HH * DD);
  unsigned short* ew3b = aus((size_t)QQ * HH);
  unsigned short* rw1b = aus((size_t)DD * DD);
  unsigned short* rw2b = aus((size_t)HH * DD);
  unsigned short* rw3b = aus((size_t)AA * HH);
  unsigned short* owb  = aus((size_t)DD * DD);
  unsigned short* H1b  = aus((size_t)M * DD);
  unsigned short* H2b  = aus((size_t)M * HH);
  unsigned short* H3b  = aus((size_t)M * QQ);
  unsigned short* R1b  = aus((size_t)M * DD);
  unsigned short* R2b  = aus((size_t)M * HH);
  unsigned short* RXb  = aus((size_t)M * DD);
  float* CB = af32((size_t)AA * DD);
  float* TV = af32(1);

  auto cast = [&](const float* s, unsigned short* d, int rows, int cols, int ld) {
    int blocks = (int)(((size_t)rows * cols / 4 + 255) / 256);
    cast2d_bf16<<<blocks, 256, 0, stream>>>(s, d, rows, cols, ld);
  };
  cast(x, xb, M, DD, DD);
  cast(aW, aWb, AA * DD, DD, DD);
  cast(e_w1, ew1b, DD, DD, DD);
  cast(e_w2, ew2b, HH, DD, DD);
  cast(e_w3, ew3b, QQ, HH, HH);
  cast(r_w1, rw1b, DD, DD, DD + 1);  // drop concat column (kept in f32 rank-1)
  cast(r_w2, rw2b, HH, DD, DD);
  cast(r_w3, rw3b, AA, HH, HH);
  cast(o_w, owb, DD, DD, DD);

  // Energy net
  gemm_fused<DD, DD, 1, false><<<nb, 256, 0, stream>>>(xb, ew1b, e_b1, e_g1, e_be1, nullptr, nullptr, 0, H1b);
  gemm_fused<HH, DD, 1, false><<<nb, 256, 0, stream>>>(H1b, ew2b, e_b2, e_g2, e_be2, nullptr, nullptr, 0, H2b);
  gemm_fused<QQ, HH, 2, false><<<nb, 256, 0, stream>>>(H2b, ew3b, e_b3, nullptr, nullptr, nullptr, nullptr, 0, H3b);
  energy_head<<<(M + 255) / 256, 256, 0, stream>>>(H3b, e_w4, e_b4, energy, M);
  temp_reduce<<<1, 1024, 0, stream>>>(energy, TV, M);

  // Routing net (K=513 concat folded in as rank-1 epilogue from f32 r_w1)
  gemm_fused<DD, DD, 3, false><<<nb, 256, 0, stream>>>(xb, rw1b, r_b1, r_g1, r_be1, energy, r_w1, DD + 1, R1b);
  gemm_fused<HH, DD, 1, false><<<nb, 256, 0, stream>>>(R1b, rw2b, r_b2, r_g2, r_be2, nullptr, nullptr, 0, R2b);
  probs_softmax<<<nb, 128, 0, stream>>>(R2b, rw3b, r_b3, TV, probs);

  // Anchor mixture + residual, then output projection (f32 out)
  anchor_const<<<(AA * DD + 255) / 256, 256, 0, stream>>>(aW, aemb, ab, CB);
  routed_fused<<<nb, 256, 0, stream>>>(xb, x, aWb, probs, CB, RXb);
  gemm_fused<DD, DD, 0, true><<<nb, 256, 0, stream>>>(RXb, owb, o_b, nullptr, nullptr, nullptr, nullptr, 0, out);
}
